// MoEMusicBridgeN2_58583353917771
// MI455X (gfx1250) — compile-verified, matrix operand order checked by reading, not run on hardware
//
#include <hip/hip_runtime.h>
#include <math.h>

// ---------------------------------------------------------------- types
typedef __attribute__((ext_vector_type(16))) __bf16 v16bf;
typedef __attribute__((ext_vector_type(8)))  __bf16 v8bf;
typedef __attribute__((ext_vector_type(8)))  float  v8f;

static __device__ inline v16bf mk16(v8bf lo, v8bf hi) {
  union { v16bf f; v8bf h[2]; } u;
  u.h[0] = lo; u.h[1] = hi;
  return u.f;
}

static __device__ inline v8f wmma_bf16(v16bf a, v16bf b, v8f c) {
  // D = A(16x32) * B(32x16) + C, f32 accumulate
  return __builtin_amdgcn_wmma_f32_16x16x32_bf16(false, a, false, b, (short)0, c,
                                                 false, false);
}

// ---------------------------------------------------------------- GEMM
// C[M,N] = A[M,K] @ W[N,K]^T (+bias[N]) (+residual[M,N])
// fp32 in global, converted to bf16 while staging into LDS.
// Block: 256 threads (8 waves). Block tile 128x128, K-step 32.
// Wave w -> (wm = w>>1, wn = w&1): 32x64 tile = 2x4 WMMA C tiles.
// FULL=true: all dims tile-aligned -> no bounds logic anywhere (hot path).
// Register double-buffer: next K-tile is fetched while WMMAs consume LDS.
template <bool FULL>
__global__ __launch_bounds__(256) void gemm_bf16_kernel(
    const float* __restrict__ A, int lda,
    const float* __restrict__ W, int ldw,
    float* __restrict__ C, int ldc,
    const float* __restrict__ bias,
    const float* __restrict__ residual, int ldr,
    int M, int N, int K)
{
  __shared__ __bf16 As[128][40];   // pad 32->40 to stagger LDS banks
  __shared__ __bf16 Bs[128][40];

  const int tid  = threadIdx.x;
  const int lane = tid & 31;
  const int w    = tid >> 5;
  const int wm   = w >> 1;         // 0..3
  const int wn   = w & 1;          // 0..1
  const int m0   = blockIdx.y * 128;
  const int n0   = blockIdx.x * 128;

  const int lr    = lane & 15;
  const int half  = lane >> 4;
  const int kbase = half * 8;      // 16-bit A/B fragment K base per lane half

  int rowi[4], coli[4];
#pragma unroll
  for (int i = 0; i < 4; ++i) {
    const int f = tid + i * 256;   // float4 index 0..1023 of the 128x32 tile
    rowi[i] = f >> 3;
    coli[i] = (f & 7) << 2;
  }

  v8f acc[2][4];
#pragma unroll
  for (int i = 0; i < 2; ++i)
#pragma unroll
    for (int j = 0; j < 4; ++j)
      acc[i][j] = (v8f){0.f,0.f,0.f,0.f,0.f,0.f,0.f,0.f};

  float4 ra[4], rw[4];

  // batched fetch of one 128x32 A tile + one 128x32 W tile
  auto fetch = [&](int k0) {
#pragma unroll
    for (int i = 0; i < 4; ++i) {
      if (FULL) {
        const int gc = k0 + coli[i];
        ra[i] = *(const float4*)(A + (size_t)(m0 + rowi[i]) * lda + gc);
        rw[i] = *(const float4*)(W + (size_t)(n0 + rowi[i]) * ldw + gc);
        if (k0 + 64 < K) {          // warm L2 two tiles ahead
          __builtin_prefetch(A + (size_t)(m0 + rowi[i]) * lda + gc + 64, 0, 1);
          __builtin_prefetch(W + (size_t)(n0 + rowi[i]) * ldw + gc + 64, 0, 1);
        }
      } else {
        const int gc   = k0 + coli[i];
        const bool kok = gc < K;
        const int gck  = kok ? gc : (K - 4);        // K is always %4 == 0

        const int ar   = m0 + rowi[i];
        const bool aok = kok && (ar < M);
        const int arc  = (ar < M) ? ar : (M - 1);
        const float4 va = *(const float4*)(A + (size_t)arc * lda + gck);
        ra[i] = aok ? va : make_float4(0.f, 0.f, 0.f, 0.f);

        const int wr   = n0 + rowi[i];
        const bool wok = kok && (wr < N);
        const int wrc  = (wr < N) ? wr : (N - 1);
        const float4 vw = *(const float4*)(W + (size_t)wrc * ldw + gck);
        rw[i] = wok ? vw : make_float4(0.f, 0.f, 0.f, 0.f);
      }
    }
  };

  // fp32 -> bf16 convert + LDS store of the register-held tiles
  auto stage = [&]() {
#pragma unroll
    for (int i = 0; i < 4; ++i) {
      __bf16* da = &As[rowi[i]][coli[i]];
      da[0] = (__bf16)ra[i].x; da[1] = (__bf16)ra[i].y;
      da[2] = (__bf16)ra[i].z; da[3] = (__bf16)ra[i].w;
      __bf16* db = &Bs[rowi[i]][coli[i]];
      db[0] = (__bf16)rw[i].x; db[1] = (__bf16)rw[i].y;
      db[2] = (__bf16)rw[i].z; db[3] = (__bf16)rw[i].w;
    }
  };

  fetch(0);
  for (int k0 = 0; k0 < K; k0 += 32) {
    stage();
    __syncthreads();
    if (k0 + 32 < K) fetch(k0 + 32);   // loads fly while WMMAs run below

    v16bf af[2], bfr[4];
#pragma unroll
    for (int mi = 0; mi < 2; ++mi) {
      const int r = wm * 32 + mi * 16 + lr;
      af[mi] = mk16(*(const v8bf*)&As[r][kbase],
                    *(const v8bf*)&As[r][kbase + 16]);
    }
#pragma unroll
    for (int ni = 0; ni < 4; ++ni) {
      const int r = wn * 64 + ni * 16 + lr;
      bfr[ni] = mk16(*(const v8bf*)&Bs[r][kbase],
                     *(const v8bf*)&Bs[r][kbase + 16]);
    }
#pragma unroll
    for (int mi = 0; mi < 2; ++mi)
#pragma unroll
      for (int ni = 0; ni < 4; ++ni)
        acc[mi][ni] = wmma_bf16(af[mi], bfr[ni], acc[mi][ni]);
    __syncthreads();
  }

  // ---- epilogue: bias + residual + store
#pragma unroll
  for (int mi = 0; mi < 2; ++mi) {
#pragma unroll
    for (int ni = 0; ni < 4; ++ni) {
      const int gcol = n0 + wn * 64 + ni * 16 + lr;
      float bv = 0.f;
      if (bias && (FULL || gcol < N)) bv = bias[gcol];
#pragma unroll
      for (int j = 0; j < 8; ++j) {
        const int grow = m0 + wm * 32 + mi * 16 + j + half * 8;
        if (FULL || (grow < M && gcol < N)) {
          float v = acc[mi][ni][j] + bv;
          if (residual) v += residual[(size_t)grow * ldr + gcol];
          C[(size_t)grow * ldc + gcol] = v;
        }
      }
    }
  }
}

// ---------------------------------------------------------------- converts
__global__ void cvt_bf16_kernel(const float* __restrict__ in,
                                __bf16* __restrict__ out, long n) {
  for (long i = (long)blockIdx.x * blockDim.x + threadIdx.x; i < n;
       i += (long)gridDim.x * blockDim.x)
    out[i] = (__bf16)in[i];
}

// vt[head*256*2048 + d*2048 + t] = qkv[t*6144 + 4096 + head*256 + d]
__global__ void build_vt_kernel(const float* __restrict__ qkv,
                                __bf16* __restrict__ vt) {
  const long i = (long)blockIdx.x * blockDim.x + threadIdx.x;
  if (i >= (long)8 * 256 * 2048) return;
  const int t  = (int)(i & 2047);
  const int hd = (int)(i >> 11);              // head*256 + d
  vt[i] = (__bf16)qkv[(size_t)t * 6144 + 4096 + hd];
}

// ---------------------------------------------------------------- flash attn
// grid (16, 8): 16 q-blocks of 128 rows x 8 heads. 256 threads = 8 waves,
// each wave owns a 16-row q tile, streams keys in 32-wide chunks.
__global__ __launch_bounds__(256) void flash_attn_kernel(
    const __bf16* __restrict__ qkvh,   // 2048 x 6144 bf16 (q|k|v)
    const __bf16* __restrict__ vt,     // 8 x 256 x 2048 bf16 (V^T per head)
    float* __restrict__ out,           // 2048 x 2048 fp32
    int T)
{
  __shared__ __bf16 Pl[8][16][40];     // per-wave P staging (C -> A relayout)

  const int tid  = threadIdx.x;
  const int lane = tid & 31;
  const int w    = tid >> 5;
  const int head = blockIdx.y;
  const int q0   = blockIdx.x * 128 + w * 16;
  const int lr   = lane & 15;
  const int half = lane >> 4;
  const int kb8  = half * 8;
  const size_t LD = 6144;

  // Q fragments (16 x 256, 8 K-chunks of 32) -- kept resident
  v16bf qf[8];
  {
    const __bf16* qrow = qkvh + (size_t)(q0 + lr) * LD + head * 256;
#pragma unroll
    for (int kd = 0; kd < 8; ++kd) {
      const __bf16* p = qrow + kd * 32 + kb8;
      qf[kd] = mk16(*(const v8bf*)p, *(const v8bf*)(p + 16));
    }
  }

  v8f o[16];
#pragma unroll
  for (int i = 0; i < 16; ++i) o[i] = (v8f){0.f,0.f,0.f,0.f,0.f,0.f,0.f,0.f};
  float mrun[8], lrun[8], alpha[8];
#pragma unroll
  for (int j = 0; j < 8; ++j) { mrun[j] = -3.0e38f; lrun[j] = 0.f; }

  const float scale = 0.0625f;  // 1/sqrt(256)

  for (int kc = 0; kc < T / 32; ++kc) {
    const int key0 = kc * 32;

    // S = Q K^T : two 16x16 C tiles (keys 0..15 / 16..31 of chunk)
    v8f s0 = (v8f){0.f,0.f,0.f,0.f,0.f,0.f,0.f,0.f};
    v8f s1 = (v8f){0.f,0.f,0.f,0.f,0.f,0.f,0.f,0.f};
#pragma unroll
    for (int kd = 0; kd < 8; ++kd) {
      const __bf16* k0p = qkvh + (size_t)(key0 + lr) * LD + 2048 + head * 256 +
                          kd * 32 + kb8;
      const __bf16* k1p = qkvh + (size_t)(key0 + 16 + lr) * LD + 2048 +
                          head * 256 + kd * 32 + kb8;
      v16bf b0f = mk16(*(const v8bf*)k0p, *(const v8bf*)(k0p + 16));
      v16bf b1f = mk16(*(const v8bf*)k1p, *(const v8bf*)(k1p + 16));
      s0 = wmma_bf16(qf[kd], b0f, s0);
      s1 = wmma_bf16(qf[kd], b1f, s1);
    }

    // online softmax per row (row j + 8*half lives across a 16-lane group)
#pragma unroll
    for (int j = 0; j < 8; ++j) {
      float a0 = s0[j] * scale, a1 = s1[j] * scale;
      float mx = fmaxf(a0, a1);
#pragma unroll
      for (int m = 1; m <= 8; m <<= 1) mx = fmaxf(mx, __shfl_xor(mx, m, 32));
      const float nm = fmaxf(mrun[j], mx);
      const float p0 = __expf(a0 - nm);
      const float p1 = __expf(a1 - nm);
      float rs = p0 + p1;
#pragma unroll
      for (int m = 1; m <= 8; m <<= 1) rs += __shfl_xor(rs, m, 32);
      const float al = __expf(mrun[j] - nm);
      mrun[j]  = nm;
      lrun[j]  = lrun[j] * al + rs;
      alpha[j] = al;
      // park P in LDS in C-layout
      Pl[w][j + half * 8][lr]      = (__bf16)p0;
      Pl[w][j + half * 8][16 + lr] = (__bf16)p1;
    }

    // rescale running O
#pragma unroll
    for (int ni = 0; ni < 16; ++ni)
#pragma unroll
      for (int j = 0; j < 8; ++j) o[ni][j] *= alpha[j];

    asm volatile("s_wait_dscnt 0x0" ::: "memory");

    // reload P as a 16x32 A fragment
    const __bf16* pr = &Pl[w][lr][kb8];
    v16bf pf = mk16(*(const v8bf*)pr, *(const v8bf*)(pr + 16));

    // O += P @ V  (B fragments from V^T: contiguous along keys)
    const __bf16* vhead = vt + (size_t)head * 256 * 2048;
#pragma unroll
    for (int ni = 0; ni < 16; ++ni) {
      const __bf16* vr = vhead + (size_t)(ni * 16 + lr) * 2048 + key0 + kb8;
      v16bf bv = mk16(*(const v8bf*)vr, *(const v8bf*)(vr + 16));
      o[ni] = wmma_bf16(pf, bv, o[ni]);
    }
  }

  // normalize + write (head-interleaved layout: col = head*256 + d)
#pragma unroll
  for (int ni = 0; ni < 16; ++ni) {
    const int col = head * 256 + ni * 16 + lr;
#pragma unroll
    for (int j = 0; j < 8; ++j) {
      const int row = q0 + j + half * 8;
      out[(size_t)row * 2048 + col] = o[ni][j] / lrun[j];
    }
  }
}

// ---------------------------------------------------------------- layernorm
__global__ __launch_bounds__(256) void layernorm_kernel(
    const float* __restrict__ x, const float* __restrict__ g,
    const float* __restrict__ b, float* __restrict__ y, int D)
{
  const int row = blockIdx.x;
  const float* xr = x + (size_t)row * D;
  float* yr = y + (size_t)row * D;
  float s = 0.f, s2 = 0.f;
  for (int i = threadIdx.x; i < D; i += 256) {
    const float v = xr[i];
    s += v; s2 += v * v;
  }
#pragma unroll
  for (int m = 16; m; m >>= 1) {
    s  += __shfl_xor(s, m, 32);
    s2 += __shfl_xor(s2, m, 32);
  }
  __shared__ float ssum[8], ssq[8];
  const int w = threadIdx.x >> 5;
  if ((threadIdx.x & 31) == 0) { ssum[w] = s; ssq[w] = s2; }
  __syncthreads();
  if (threadIdx.x == 0) {
    float a = 0.f, c = 0.f;
    for (int i = 0; i < 8; ++i) { a += ssum[i]; c += ssq[i]; }
    const float mu = a / D;
    ssum[0] = mu;
    ssq[0]  = rsqrtf(c / D - mu * mu + 1e-5f);
  }
  __syncthreads();
  const float mu = ssum[0], rs = ssq[0];
  for (int i = threadIdx.x; i < D; i += 256)
    yr[i] = (xr[i] - mu) * rs * g[i] + b[i];
}

// ---------------------------------------------------------------- conv+silu
// xc[t,c] = silu( sum_j w[j,0,c]*xz[t-3+j, c] + b[c] ), xz row stride 8192
__global__ void conv_silu_kernel(const float* __restrict__ xz,
                                 const float* __restrict__ w,
                                 const float* __restrict__ b,
                                 float* __restrict__ out, int T)
{
  const long i = (long)blockIdx.x * blockDim.x + threadIdx.x;
  if (i >= (long)T * 4096) return;
  const int c = (int)(i & 4095);
  const int t = (int)(i >> 12);
  float acc = b[c];
#pragma unroll
  for (int j = 0; j < 4; ++j) {
    const int ts = t - 3 + j;
    if (ts >= 0) acc += w[j * 4096 + c] * xz[(size_t)ts * 8192 + c];
  }
  out[(size_t)t * 4096 + c] = acc / (1.f + __expf(-acc));
}

// ---------------------------------------------------------------- mamba scan
// thread = (channel c, state s); 16 state-lanes reduce y via shfl_xor.
__global__ __launch_bounds__(256) void mamba_scan_kernel(
    const float* __restrict__ xdbl,   // 2048 x 160 (dt|B|C)
    const float* __restrict__ dtpre,  // 2048 x 4096 (pre-softplus, +bdt)
    const float* __restrict__ xc,     // 2048 x 4096
    const float* __restrict__ xz,     // 2048 x 8192 (z at +4096)
    const float* __restrict__ A_log,  // 4096 x 16
    const float* __restrict__ Dp,     // 4096
    float* __restrict__ y,            // 2048 x 4096
    int T)
{
  const int s = threadIdx.x & 15;
  const int c = blockIdx.x * 16 + (threadIdx.x >> 4);
  const float a  = -__expf(A_log[(size_t)c * 16 + s]);
  const float Dc = Dp[c];
  float h = 0.f;
  for (int t = 0; t < T; ++t) {
    const float dp = dtpre[(size_t)t * 4096 + c];
    const float dt = (dp > 20.f) ? dp : log1pf(__expf(dp));
    const float xv = xc[(size_t)t * 4096 + c];
    const float bm = xdbl[(size_t)t * 160 + 128 + s];
    const float cm = xdbl[(size_t)t * 160 + 144 + s];
    h = __expf(dt * a) * h + dt * bm * xv;
    float part = h * cm;
#pragma unroll
    for (int m = 1; m <= 8; m <<= 1) part += __shfl_xor(part, m, 32);
    if (s == 0) {
      const float zv = xz[(size_t)t * 8192 + 4096 + c];
      y[(size_t)t * 4096 + c] = (part + Dc * xv) * (zv / (1.f + __expf(-zv)));
    }
  }
}

// ---------------------------------------------------------------- bandpass
// c[d] = (1 + 2*sum_{f=1..511} cos(2 pi f d / 2048)) / 2048
__global__ void band_c_kernel(float* __restrict__ cv, int T) {
  const int d = blockIdx.x * blockDim.x + threadIdx.x;
  if (d >= T) return;
  float acc = 1.f;
  const float k = 6.28318530717958647692f * (float)d / (float)T;
  for (int f = 1; f < 512; ++f) acc += 2.f * __cosf(k * (float)f);
  cv[d] = acc / (float)T;
}

// Cc[t,s] = cv[(t-s) mod T]  (circulant matrix for the low band)
__global__ void build_circ_kernel(const float* __restrict__ cv,
                                  float* __restrict__ Cc, int T) {
  const long i = (long)blockIdx.x * blockDim.x + threadIdx.x;
  if (i >= (long)T * T) return;
  const int t = (int)(i >> 11);
  const int s = (int)(i & (T - 1));
  Cc[i] = cv[(t - s) & (T - 1)];
}

// dT[ch, s] = x[s*1024 + 760 + ch]  (dance transposed, 264 x 2048)
__global__ void build_danceT_kernel(const float* __restrict__ x,
                                    float* __restrict__ dT, int T) {
  const long i = (long)blockIdx.x * blockDim.x + threadIdx.x;
  if (i >= (long)264 * T) return;
  const int ch = (int)(i >> 11);
  const int s  = (int)(i & (T - 1));
  dT[i] = x[(size_t)s * 1024 + 760 + ch];
}

// b1 = dance - b0
__global__ void band1_kernel(const float* __restrict__ x,
                             const float* __restrict__ b0,
                             float* __restrict__ b1, int T) {
  const long i = (long)blockIdx.x * blockDim.x + threadIdx.x;
  if (i >= (long)T * 264) return;
  const int t  = (int)(i / 264);
  const int ch = (int)(i % 264);
  b1[i] = x[(size_t)t * 1024 + 760 + ch] - b0[i];
}

// ---------------------------------------------------------------- host side
static void run_gemm(hipStream_t st, const float* A, int lda, const float* W,
                     int ldw, float* C, int ldc, const float* bias,
                     const float* resid, int ldr, int M, int N, int K)
{
  dim3 g((N + 127) / 128, (M + 127) / 128), b(256);
  const bool full = (M % 128 == 0) && (N % 128 == 0) && (K % 32 == 0);
  if (full)
    gemm_bf16_kernel<true><<<g, b, 0, st>>>(A, lda, W, ldw, C, ldc, bias,
                                            resid, ldr, M, N, K);
  else
    gemm_bf16_kernel<false><<<g, b, 0, st>>>(A, lda, W, ldw, C, ldc, bias,
                                             resid, ldr, M, N, K);
}

static void run_expert(hipStream_t st, const float* X, int ldx, int din,
                       void* const* d_in, int base, float* catBase, int slot,
                       float* hbuf, float* qkv, __bf16* qkvh, __bf16* vt,
                       float* attn, float* abuf, float* lnbuf, float* xzb,
                       float* xcb, float* xdb, float* dtb, float* yb)
{
  const float* Wp    = (const float*)d_in[base + 0];
  const float* bp    = (const float*)d_in[base + 1];
  const float* Wqkv  = (const float*)d_in[base + 2];
  const float* bqkv  = (const float*)d_in[base + 3];
  const float* Wo    = (const float*)d_in[base + 4];
  const float* bo    = (const float*)d_in[base + 5];
  const float* lng   = (const float*)d_in[base + 6];
  const float* lnbv  = (const float*)d_in[base + 7];
  const float* Win   = (const float*)d_in[base + 8];
  const float* convw = (const float*)d_in[base + 9];
  const float* convb = (const float*)d_in[base + 10];
  const float* Wx    = (const float*)d_in[base + 11];
  const float* Wdt   = (const float*)d_in[base + 12];
  const float* bdt   = (const float*)d_in[base + 13];
  const float* Alog  = (const float*)d_in[base + 14];
  const float* Dp    = (const float*)d_in[base + 15];
  const float* Wout  = (const float*)d_in[base + 16];

  const int T = 2048;
  // h = X @ Wp^T + bp
  run_gemm(st, X, ldx, Wp, din, hbuf, 2048, bp, nullptr, 0, T, 2048, din);
  // qkv
  run_gemm(st, hbuf, 2048, Wqkv, 2048, qkv, 6144, bqkv, nullptr, 0, T, 6144, 2048);
  cvt_bf16_kernel<<<4096, 256, 0, st>>>(qkv, qkvh, (long)T * 6144);
  build_vt_kernel<<<(8 * 256 * 2048 + 255) / 256, 256, 0, st>>>(qkv, vt);
  flash_attn_kernel<<<dim3(16, 8), 256, 0, st>>>(qkvh, vt, attn, T);
  // a = attn @ Wo^T + bo
  run_gemm(st, attn, 2048, Wo, 2048, abuf, 2048, bo, nullptr, 0, T, 2048, 2048);
  layernorm_kernel<<<T, 256, 0, st>>>(abuf, lng, lnbv, lnbuf, 2048);
  // xz = ln @ Win^T
  run_gemm(st, lnbuf, 2048, Win, 2048, xzb, 8192, nullptr, nullptr, 0, T, 8192, 2048);
  conv_silu_kernel<<<(T * 4096 + 255) / 256, 256, 0, st>>>(xzb, convw, convb, xcb, T);
  // x_dbl = xc @ Wx^T
  run_gemm(st, xcb, 4096, Wx, 4096, xdb, 160, nullptr, nullptr, 0, T, 160, 4096);
  // dt_pre = x_dbl[:, :128] @ Wdt^T + bdt
  run_gemm(st, xdb, 160, Wdt, 128, dtb, 4096, bdt, nullptr, 0, T, 4096, 128);
  mamba_scan_kernel<<<256, 256, 0, st>>>(xdb, dtb, xcb, xzb, Alog, Dp, yb, T);
  // m + a -> cat slot
  run_gemm(st, yb, 4096, Wout, 4096, catBase + (size_t)slot * 2048, 6144,
           nullptr, abuf, 2048, T, 2048, 4096);
}

extern "C" void kernel_launch(void* const* d_in, const int* in_sizes, int n_in,
                              void* d_out, int out_size, void* d_ws,
                              size_t ws_size, hipStream_t stream)
{
  (void)in_sizes; (void)n_in; (void)out_size; (void)ws_size;
  const float* x   = (const float*)d_in[0];
  const float* Wip = (const float*)d_in[1];
  const float* bip = (const float*)d_in[2];
  const float* Wf  = (const float*)d_in[54];
  const float* bfu = (const float*)d_in[55];
  float* out = (float*)d_out;

  // ---- workspace carve (256B aligned)
  char* p = (char*)d_ws;
  auto take = [&](size_t bytes) -> void* {
    void* r = (void*)p;
    p += (bytes + 255) & ~(size_t)255;
    return r;
  };
  const size_t T = 2048;
  float*  xp    = (float*)take(T * 2048 * 4);
  float*  hbuf  = (float*)take(T * 2048 * 4);
  float*  qkv   = (float*)take(T * 6144 * 4);
  __bf16* qkvh  = (__bf16*)take(T * 6144 * 2);
  __bf16* vt    = (__bf16*)take((size_t)8 * 256 * 2048 * 2);
  float*  attn  = (float*)take(T * 2048 * 4);
  float*  abuf  = (float*)take(T * 2048 * 4);
  float*  lnbuf = (float*)take(T * 2048 * 4);
  float*  xzb   = (float*)take(T * 8192 * 4);
  float*  xcb   = (float*)take(T * 4096 * 4);
  float*  xdb   = (float*)take(T * 160 * 4);
  float*  dtb   = (float*)take(T * 4096 * 4);
  float*  yb    = (float*)take(T * 4096 * 4);
  float*  cat   = (float*)take(T * 6144 * 4);
  float*  b0    = (float*)take(T * 264 * 4);
  float*  b1    = (float*)take(T * 264 * 4);
  float*  cv    = (float*)take(T * 4);
  float*  circ  = (float*)take(T * T * 4);
  float*  dT    = (float*)take((size_t)264 * T * 4);

  // input projection: xp = x @ Wip^T + bip
  run_gemm(stream, x, 1024, Wip, 1024, xp, 2048, bip, nullptr, 0, T, 2048, 1024);

  // universal expert -> cat[:, 0:2048]
  run_expert(stream, xp, 2048, 2048, d_in, 3, cat, 0, hbuf, qkv, qkvh, vt,
             attn, abuf, lnbuf, xzb, xcb, xdb, dtb, yb);

  // bandpass as circulant matmul: b0 = Circ(c) @ dance, b1 = dance - b0
  band_c_kernel<<<8, 256, 0, stream>>>(cv, (int)T);
  build_circ_kernel<<<(int)((T * T + 255) / 256), 256, 0, stream>>>(cv, circ, (int)T);
  build_danceT_kernel<<<(int)((264 * T + 255) / 256), 256, 0, stream>>>(x, dT, (int)T);
  run_gemm(stream, circ, (int)T, dT, (int)T, b0, 264, nullptr, nullptr, 0,
           (int)T, 264, (int)T);
  band1_kernel<<<(int)((T * 264 + 255) / 256), 256, 0, stream>>>(x, b0, b1, (int)T);

  // spectral experts -> cat[:, 2048:4096], cat[:, 4096:6144]
  run_expert(stream, b0, 264, 264, d_in, 20, cat, 1, hbuf, qkv, qkvh, vt,
             attn, abuf, lnbuf, xzb, xcb, xdb, dtb, yb);
  run_expert(stream, b1, 264, 264, d_in, 37, cat, 2, hbuf, qkv, qkvh, vt,
             attn, abuf, lnbuf, xzb, xcb, xdb, dtb, yb);

  // fusion: out = cat @ Wf^T + bfu
  run_gemm(stream, cat, 6144, Wf, 6144, out, 2048, bfu, nullptr, 0, T, 2048, 6144);
}
